// VideoEncoder_55052890800800
// MI455X (gfx1250) — matrix operand order repaired and checked by reading, not compiled, not run on hardware
//
#include <hip/hip_runtime.h>
#include <cstdint>
#include <cstddef>

// ---------------------------------------------------------------------------
// 6-layer transformer encoder forward (B=4,S=512,D=1024,DFF=4096,H=16,DK=64)
// All GEMMs on v_wmma_f32_16x16x32_bf16 (CDNA5 wave32 WMMA), fp32 accumulate.
// GEMM tiles staged to LDS via async global->LDS copies (ASYNCcnt) when the
// toolchain exposes them, double-buffered to overlap fetch with WMMA.
// ---------------------------------------------------------------------------

#define LNUM 6
#define D    1024
#define DFF  4096
#define H    16
#define BB   4
#define S    512
#define DK   64
#define MTOK (BB * S)   // 2048 tokens
#define EPSV 1e-6f

typedef __attribute__((ext_vector_type(16))) __bf16 v16bf;
typedef __attribute__((ext_vector_type(8)))  __bf16 v8bf;
typedef __attribute__((ext_vector_type(8)))  float  v8f;
typedef __attribute__((ext_vector_type(4)))  int    v4i;

#if __has_builtin(__builtin_amdgcn_global_load_async_to_lds_b128) && \
    __has_builtin(__builtin_amdgcn_s_wait_asynccnt)
#define USE_ASYNC 1
#else
#define USE_ASYNC 0
#endif

// fp32 -> bf16 (round to nearest even)
__device__ __forceinline__ uint16_t f2bf(float f) {
    union { float f; uint32_t u; } c; c.f = f;
    uint32_t u = c.u;
    u += 0x7FFFu + ((u >> 16) & 1u);
    return (uint16_t)(u >> 16);
}

// assemble a 16-element bf16 fragment from two 16-byte chunks
__device__ __forceinline__ v16bf frag16(const uint16_t* p0, const uint16_t* p1) {
    v8bf a = *(const v8bf*)p0;
    v8bf b = *(const v8bf*)p1;
    return __builtin_shufflevector(a, b, 0,1,2,3,4,5,6,7,8,9,10,11,12,13,14,15);
}

__device__ __forceinline__ v8f wmma_bf16(v16bf a, v16bf b, v8f c) {
    // D = A(16x32 bf16) * B(32x16 bf16) + C(16x16 f32)
    return __builtin_amdgcn_wmma_f32_16x16x32_bf16(false, a, false, b, (short)0, c,
                                                   false, false);
}

#if USE_ASYNC
// 16B async copy global -> LDS (tracked by ASYNCcnt, writes LDS directly)
__device__ __forceinline__ void async_cp16B(uint16_t* lds, const uint16_t* g) {
    __builtin_amdgcn_global_load_async_to_lds_b128(
        (__attribute__((address_space(1))) v4i*)g,
        (__attribute__((address_space(3))) v4i*)lds,
        0, 0);
}
#endif

// ---------------------------------------------------------------------------
// Weight transpose + convert: fp32 [K][N] -> bf16 [N][K]
// ---------------------------------------------------------------------------
__global__ __launch_bounds__(256)
void transpose_w_kernel(const float* __restrict__ in, uint16_t* __restrict__ out,
                        int K, int N) {
    __shared__ uint16_t tile[32][33];
    int n0 = blockIdx.x * 32, k0 = blockIdx.y * 32;
    int tx = threadIdx.x & 31, ty = threadIdx.x >> 5;  // ty: 0..7
    #pragma unroll
    for (int i = 0; i < 4; ++i) {
        int k = ty + i * 8;
        tile[k][tx] = f2bf(in[(size_t)(k0 + k) * N + n0 + tx]);
    }
    __syncthreads();
    #pragma unroll
    for (int i = 0; i < 4; ++i) {
        int n = ty + i * 8;
        out[(size_t)(n0 + n) * K + k0 + tx] = tile[tx][n];
    }
}

// ---------------------------------------------------------------------------
// V transpose: v bf16 [B*S][D] (per-head cols) -> vT bf16 [B*H][DK][S]
// ---------------------------------------------------------------------------
__global__ __launch_bounds__(256)
void transpose_v_kernel(const uint16_t* __restrict__ v, uint16_t* __restrict__ vt) {
    __shared__ uint16_t tile[32][33];
    int bh = blockIdx.z;
    int b = bh / H, h = bh % H;
    int d0 = blockIdx.x * 32, s0 = blockIdx.y * 32;
    int tx = threadIdx.x & 31, ty = threadIdx.x >> 5;
    #pragma unroll
    for (int i = 0; i < 4; ++i) {
        int s = ty + i * 8;
        tile[s][tx] = v[(size_t)(b * S + s0 + s) * D + h * DK + d0 + tx];
    }
    __syncthreads();
    #pragma unroll
    for (int i = 0; i < 4; ++i) {
        int d = ty + i * 8;
        vt[((size_t)bh * DK + d0 + d) * S + s0 + tx] = tile[tx][d];
    }
}

// ---------------------------------------------------------------------------
// LayerNorm (sample std, ddof=1, eps added to std) over D=1024.
// One row per block, 256 threads * 4 elements. OUTF=0 -> bf16, 1 -> fp32.
// ---------------------------------------------------------------------------
template<int OUTF>
__global__ __launch_bounds__(256)
void layernorm_kernel(const float* __restrict__ x, const float* __restrict__ ga,
                      const float* __restrict__ gb, uint16_t* __restrict__ outb,
                      float* __restrict__ outf) {
    int row = blockIdx.x;
    const float* xr = x + (size_t)row * D;
    int t = threadIdx.x;
    float4 v4 = ((const float4*)xr)[t];
    float vv[4] = {v4.x, v4.y, v4.z, v4.w};
    float s = vv[0] + vv[1] + vv[2] + vv[3];
    float q = vv[0]*vv[0] + vv[1]*vv[1] + vv[2]*vv[2] + vv[3]*vv[3];
    #pragma unroll
    for (int off = 16; off; off >>= 1) {
        s += __shfl_xor(s, off);
        q += __shfl_xor(q, off);
    }
    __shared__ float ssum[8], ssq[8];
    int wid = t >> 5, lid = t & 31;
    if (lid == 0) { ssum[wid] = s; ssq[wid] = q; }
    __syncthreads();
    float S8 = 0.f, Q8 = 0.f;
    #pragma unroll
    for (int i = 0; i < 8; ++i) { S8 += ssum[i]; Q8 += ssq[i]; }
    float mean = S8 * (1.0f / D);
    float var  = (Q8 - (float)D * mean * mean) * (1.0f / (D - 1));
    var = var > 0.f ? var : 0.f;
    float inv = 1.0f / (sqrtf(var) + EPSV);
    #pragma unroll
    for (int j = 0; j < 4; ++j) {
        int c = t * 4 + j;
        float val = ga[c] * (vv[j] - mean) * inv + gb[c];
        if (OUTF) outf[(size_t)row * D + c] = val;
        else      outb[(size_t)row * D + c] = f2bf(val);
    }
}

// ---------------------------------------------------------------------------
// Tiled WMMA GEMM:  C[M][N] = A[M][K](bf16) * Bt[N][K]^T(bf16) + bias
// BM=128, BN=64, KB=32. 256 threads = 8 waves; wave -> 32x32 (2x2 WMMA tiles).
// Double-buffered LDS stages; async global->LDS copies overlap the WMMAs.
// EPI: 0 = bias -> bf16 out, 1 = bias+ReLU -> bf16 out, 2 = bias+residual -> fp32
// ---------------------------------------------------------------------------
#define BM 128
#define BN 64
#define KB 32
#define LDA 40   // uint16 stride (32 + 8 pad; 80B rows, 16B aligned chunks)
#define LDB 40

__device__ __forceinline__ void stage_tiles(uint16_t* sAbuf, uint16_t* sBbuf,
                                            const uint16_t* __restrict__ A,
                                            const uint16_t* __restrict__ Bt,
                                            int m0, int n0, int k0, int Kdim, int tid) {
    int ra = tid >> 1, ca = (tid & 1) * 16;
    const uint16_t* ga = A + (size_t)(m0 + ra) * Kdim + k0 + ca;
    uint16_t* la = sAbuf + ra * LDA + ca;
    int rb = tid >> 2, cb = (tid & 3) * 8;
    const uint16_t* gb = Bt + (size_t)(n0 + rb) * Kdim + k0 + cb;
    uint16_t* lb = sBbuf + rb * LDB + cb;
#if USE_ASYNC
    async_cp16B(la,     ga);       // 3 async instructions per wave per stage
    async_cp16B(la + 8, ga + 8);
    async_cp16B(lb,     gb);
#else
    *(v8bf*)(la)     = *(const v8bf*)(ga);
    *(v8bf*)(la + 8) = *(const v8bf*)(ga + 8);
    *(v8bf*)(lb)     = *(const v8bf*)(gb);
#endif
}

template<int EPI>
__global__ __launch_bounds__(256)
void gemm_bf16_kernel(const uint16_t* __restrict__ A, const uint16_t* __restrict__ Bt,
                      const float* __restrict__ bias, float* __restrict__ X,
                      uint16_t* __restrict__ Ob, int Mdim, int Ndim, int Kdim) {
    __shared__ uint16_t sA[2][BM * LDA];
    __shared__ uint16_t sB[2][BN * LDB];
    int m0 = blockIdx.y * BM, n0 = blockIdx.x * BN;
    int tid = threadIdx.x;
    int wid = tid >> 5, lane = tid & 31;
    int wm = wid & 3, wn = wid >> 2;       // wave covers rows wm*32, cols wn*32
    int lrow = lane & 15, sel = lane >> 4;

    v8f zf = {0.f,0.f,0.f,0.f,0.f,0.f,0.f,0.f};
    v8f acc[2][2];
    acc[0][0] = zf; acc[0][1] = zf; acc[1][0] = zf; acc[1][1] = zf;

    int nk = Kdim / KB;
    stage_tiles(sA[0], sB[0], A, Bt, m0, n0, 0, Kdim, tid);

    for (int kt = 0; kt < nk; ++kt) {
        int cur = kt & 1;
        if (kt + 1 < nk) {
            // issue next stage into the other buffer (overlaps this stage's WMMAs)
            stage_tiles(sA[cur ^ 1], sB[cur ^ 1], A, Bt, m0, n0, (kt + 1) * KB, Kdim, tid);
#if USE_ASYNC
            __builtin_amdgcn_s_wait_asynccnt(3);  // current stage complete; 3 in flight
#endif
        } else {
#if USE_ASYNC
            __builtin_amdgcn_s_wait_asynccnt(0);  // drain final stage
#endif
        }
        __syncthreads();  // current stage visible to all waves

        v16bf af[2], bfr[2];
        #pragma unroll
        for (int i = 0; i < 2; ++i) {
            const uint16_t* p = sA[cur] + (wm * 32 + i * 16 + lrow) * LDA + sel * 8;
            af[i] = frag16(p, p + 16);
        }
        #pragma unroll
        for (int j = 0; j < 2; ++j) {
            const uint16_t* p = sB[cur] + (wn * 32 + j * 16 + lrow) * LDB + sel * 8;
            bfr[j] = frag16(p, p + 16);
        }
        #pragma unroll
        for (int i = 0; i < 2; ++i)
            #pragma unroll
            for (int j = 0; j < 2; ++j)
                acc[i][j] = wmma_bf16(af[i], bfr[j], acc[i][j]);

        __syncthreads();  // protect buffer reuse (restaged next iteration)
    }

    // epilogue
    #pragma unroll
    for (int i = 0; i < 2; ++i) {
        #pragma unroll
        for (int j = 0; j < 2; ++j) {
            int n = n0 + wn * 32 + j * 16 + lrow;
            float bsv = bias[n];
            int mbase = m0 + wm * 32 + i * 16 + sel * 8;
            #pragma unroll
            for (int r = 0; r < 8; ++r) {
                size_t idx = (size_t)(mbase + r) * Ndim + n;
                float val = acc[i][j][r] + bsv;
                if (EPI == 0)      Ob[idx] = f2bf(val);
                else if (EPI == 1) Ob[idx] = f2bf(val > 0.f ? val : 0.f);
                else               X[idx]  = X[idx] + val;
            }
        }
    }
}

// ---------------------------------------------------------------------------
// Fused attention per (b, h, 64-query tile): scores = q k^T / sqrt(DK),
// softmax rows in LDS, o = p v. All GEMMs via WMMA bf16.
// ---------------------------------------------------------------------------
#define QT   64
#define SCST 516   // score row stride (floats)
#define PST  520   // p row stride (uint16)
#define ATT_SMEM (QT * SCST * 4 + QT * PST * 2)

__global__ __launch_bounds__(256)
void attention_kernel(const uint16_t* __restrict__ q, const uint16_t* __restrict__ k,
                      const uint16_t* __restrict__ vt, uint16_t* __restrict__ o) {
    extern __shared__ char smem[];
    float*    sc = (float*)smem;                       // [QT][SCST]
    uint16_t* sp = (uint16_t*)(smem + QT * SCST * 4);  // [QT][PST]
    int bh = blockIdx.y;
    int b = bh / H, h = bh % H;
    int q0 = blockIdx.x * QT;
    int tid = threadIdx.x, wid = tid >> 5, lane = tid & 31;
    int lrow = lane & 15, sel = lane >> 4;
    const float inv = 0.125f;  // 1/sqrt(64)
    v8f zf = {0.f,0.f,0.f,0.f,0.f,0.f,0.f,0.f};

    // phase 1: scores (4 qtiles x 32 ktiles = 128 WMMA tiles over 8 waves)
    for (int t = 0; t < 16; ++t) {
        int tile = wid + t * 8;
        int qt = tile >> 5, kt = tile & 31;
        v8f acc = zf;
        #pragma unroll
        for (int d0 = 0; d0 < DK; d0 += 32) {
            const uint16_t* pa =
                q + (size_t)(b * S + q0 + qt * 16 + lrow) * D + h * DK + d0 + sel * 8;
            v16bf a = frag16(pa, pa + 16);
            const uint16_t* pb =
                k + (size_t)(b * S + kt * 16 + lrow) * D + h * DK + d0 + sel * 8;
            v16bf bb = frag16(pb, pb + 16);
            acc = wmma_bf16(a, bb, acc);
        }
        int n = kt * 16 + lrow;
        int mb = qt * 16 + sel * 8;
        #pragma unroll
        for (int r = 0; r < 8; ++r) sc[(mb + r) * SCST + n] = acc[r] * inv;
    }
    __syncthreads();

    // phase 2: softmax (4 threads per row, 128 cols each, shfl_xor reduce)
    {
        int r = tid >> 2, part = tid & 3;
        float* row = sc + r * SCST + part * 128;
        float mx = -1e30f;
        for (int i = 0; i < 128; ++i) mx = fmaxf(mx, row[i]);
        mx = fmaxf(mx, __shfl_xor(mx, 1));
        mx = fmaxf(mx, __shfl_xor(mx, 2));
        float sm = 0.f;
        for (int i = 0; i < 128; ++i) { float e = __expf(row[i] - mx); row[i] = e; sm += e; }
        sm += __shfl_xor(sm, 1);
        sm += __shfl_xor(sm, 2);
        float is = 1.0f / sm;
        uint16_t* prow = sp + r * PST + part * 128;
        for (int i = 0; i < 128; ++i) prow[i] = f2bf(row[i] * is);
    }
    __syncthreads();

    // phase 3: o = p @ v  (64x64 out = 16 WMMA tiles over 8 waves, K=512)
    for (int t = 0; t < 2; ++t) {
        int tile = wid + t * 8;
        int mt = tile >> 2, nt = tile & 3;
        v8f acc = zf;
        for (int k0 = 0; k0 < S; k0 += 32) {
            const uint16_t* pa = sp + (mt * 16 + lrow) * PST + k0 + sel * 8;
            v16bf a = frag16(pa, pa + 16);
            const uint16_t* pb =
                vt + ((size_t)bh * DK + nt * 16 + lrow) * S + k0 + sel * 8;
            v16bf bb = frag16(pb, pb + 16);
            acc = wmma_bf16(a, bb, acc);
        }
        int col = h * DK + nt * 16 + lrow;
        int mb = q0 + mt * 16 + sel * 8;
        #pragma unroll
        for (int r = 0; r < 8; ++r)
            o[(size_t)(b * S + mb + r) * D + col] = f2bf(acc[r]);
    }
}

// ---------------------------------------------------------------------------
// Host orchestration
// ---------------------------------------------------------------------------
extern "C" void kernel_launch(void* const* d_in, const int* in_sizes, int n_in,
                              void* d_out, int out_size, void* d_ws, size_t ws_size,
                              hipStream_t stream) {
    const float* x_in = (const float*)d_in[0];
    const float* Wq = (const float*)d_in[1];  const float* bq = (const float*)d_in[2];
    const float* Wk = (const float*)d_in[3];  const float* bk = (const float*)d_in[4];
    const float* Wv = (const float*)d_in[5];  const float* bv = (const float*)d_in[6];
    const float* Wo = (const float*)d_in[7];  const float* bo = (const float*)d_in[8];
    const float* W1 = (const float*)d_in[9];  const float* b1 = (const float*)d_in[10];
    const float* W2 = (const float*)d_in[11]; const float* b2 = (const float*)d_in[12];
    const float* l1a = (const float*)d_in[13]; const float* l1b = (const float*)d_in[14];
    const float* l2a = (const float*)d_in[15]; const float* l2b = (const float*)d_in[16];
    const float* na  = (const float*)d_in[17]; const float* nb  = (const float*)d_in[18];

    char* ws = (char*)d_ws;
    float*    X   = (float*)(ws);                        //  8 MB fp32 activations
    uint16_t* Hb  = (uint16_t*)(ws + ( 8u << 20));       //  4 MB ln output (bf16)
    uint16_t* Qb  = (uint16_t*)(ws + (12u << 20));       //  4 MB
    uint16_t* Kb  = (uint16_t*)(ws + (16u << 20));       //  4 MB
    uint16_t* Vb  = (uint16_t*)(ws + (20u << 20));       //  4 MB
    uint16_t* VT  = (uint16_t*)(ws + (24u << 20));       //  4 MB
    uint16_t* Ob  = (uint16_t*)(ws + (28u << 20));       //  4 MB
    uint16_t* Fb  = (uint16_t*)(ws + (32u << 20));       // 16 MB ffn mid (bf16)
    uint16_t* Wst = (uint16_t*)(ws + (48u << 20));       //  8 MB weight stage (bf16 [N][K])

    (void)hipMemcpyAsync(X, x_in, (size_t)MTOK * D * sizeof(float),
                         hipMemcpyDeviceToDevice, stream);

    dim3 blk(256);
    dim3 gemmDD(D / BN, MTOK / BM);
    dim3 gemmDF(DFF / BN, MTOK / BM);

    for (int l = 0; l < LNUM; ++l) {
        // --- attention sublayer ---
        layernorm_kernel<0><<<MTOK, blk, 0, stream>>>(X, l1a + l * D, l1b + l * D, Hb, nullptr);

        transpose_w_kernel<<<dim3(D/32, D/32), blk, 0, stream>>>(Wq + (size_t)l*D*D, Wst, D, D);
        gemm_bf16_kernel<0><<<gemmDD, blk, 0, stream>>>(Hb, Wst, bq + l*D, nullptr, Qb, MTOK, D, D);

        transpose_w_kernel<<<dim3(D/32, D/32), blk, 0, stream>>>(Wk + (size_t)l*D*D, Wst, D, D);
        gemm_bf16_kernel<0><<<gemmDD, blk, 0, stream>>>(Hb, Wst, bk + l*D, nullptr, Kb, MTOK, D, D);

        transpose_w_kernel<<<dim3(D/32, D/32), blk, 0, stream>>>(Wv + (size_t)l*D*D, Wst, D, D);
        gemm_bf16_kernel<0><<<gemmDD, blk, 0, stream>>>(Hb, Wst, bv + l*D, nullptr, Vb, MTOK, D, D);

        transpose_v_kernel<<<dim3(DK/32, S/32, BB*H), blk, 0, stream>>>(Vb, VT);

        attention_kernel<<<dim3(S/QT, BB*H), blk, ATT_SMEM, stream>>>(Qb, Kb, VT, Ob);

        transpose_w_kernel<<<dim3(D/32, D/32), blk, 0, stream>>>(Wo + (size_t)l*D*D, Wst, D, D);
        gemm_bf16_kernel<2><<<gemmDD, blk, 0, stream>>>(Ob, Wst, bo + l*D, X, nullptr, MTOK, D, D);

        // --- feed-forward sublayer ---
        layernorm_kernel<0><<<MTOK, blk, 0, stream>>>(X, l2a + l * D, l2b + l * D, Hb, nullptr);

        transpose_w_kernel<<<dim3(DFF/32, D/32), blk, 0, stream>>>(W1 + (size_t)l*D*DFF, Wst, D, DFF);
        gemm_bf16_kernel<1><<<gemmDF, blk, 0, stream>>>(Hb, Wst, b1 + l*DFF, nullptr, Fb, MTOK, DFF, D);

        transpose_w_kernel<<<dim3(D/32, DFF/32), blk, 0, stream>>>(W2 + (size_t)l*DFF*D, Wst, DFF, D);
        gemm_bf16_kernel<2><<<gemmDD, blk, 0, stream>>>(Fb, Wst, b2 + l*D, X, nullptr, MTOK, D, DFF);
    }

    // final LayerNorm -> fp32 output
    layernorm_kernel<1><<<MTOK, blk, 0, stream>>>(X, na, nb, nullptr, (float*)d_out);
}
